// GCN_71502615544011
// MI455X (gfx1250) — matrix operand order repaired and checked by reading, not compiled, not run on hardware
//
#include <hip/hip_runtime.h>
#include <hip/hip_bf16.h>
#include <math.h>

typedef __attribute__((ext_vector_type(16))) __bf16 v16bf;
typedef __attribute__((ext_vector_type(8)))  float  v8f;
typedef unsigned short ushort_t;

typedef unsigned int  u32x4 __attribute__((ext_vector_type(4)));
typedef int           i32x4 __attribute__((ext_vector_type(4)));
typedef int           i32x8 __attribute__((ext_vector_type(8)));

#define EPSV 1e-8f

__device__ __forceinline__ ushort_t f32_to_bf16_rte(float f) {
    unsigned int x = __float_as_uint(f);
    unsigned int r = x + 0x7FFFu + ((x >> 16) & 1u);
    return (ushort_t)(r >> 16);
}

// ---------------------------------------------------------------------------
// TDM: one descriptor loads a 64 x K bf16 tile (row-padded by 16B) into LDS.
// D# packing per CDNA5 ISA 8.3/8.4: group0 = {count/type/addrs}, group1 =
// {data_size, pad, tensor/tile dims, dim0 stride}. Rows past (M - mBase) are
// OOB reads -> zero-filled in LDS. Issued by one wave; tracked by TENSORcnt.
// ---------------------------------------------------------------------------
__device__ __forceinline__ void tdm_load_tile_bf16(const ushort_t* gsrc,
                                                   unsigned lds_off, int K_,
                                                   int rows_total, int pad_code) {
    unsigned long long ga = (unsigned long long)gsrc;
    u32x4 g0;
    g0.x = 0x1u;                                   // count = 1 (valid descriptor)
    g0.y = lds_off;                                // lds_addr (bytes)
    g0.z = (unsigned)(ga & 0xFFFFFFFFull);         // global_addr[31:0]
    g0.w = (unsigned)((ga >> 32) & 0x01FFFFFFull)  // global_addr[56:32]
         | (2u << 30);                             // type = 2 ("image")
    i32x8 g1;
    g1[0] = (1 << 16)                              // data_size = 2 bytes
          | (1 << 20)                              // pad_enable
          | (pad_code << 22)                       // pad after 2^(c+1) DWORDs (one row)
          | (3 << 25);                             // pad_amount = 4 DWORDs (16 B)
    g1[1] = (K_ & 0xFFFF) << 16;                                   // tensor_dim0[15:0]
    g1[2] = ((K_ >> 16) & 0xFFFF) | ((rows_total & 0xFFFF) << 16); // dim0 hi | dim1 lo
    g1[3] = ((rows_total >> 16) & 0xFFFF) | ((K_ & 0xFFFF) << 16); // dim1 hi | tile_dim0
    g1[4] = 64;                                    // tile_dim1 = 64 rows (tile_dim2 = 0)
    g1[5] = K_;                                    // tensor_dim0_stride[31:0]
    g1[6] = 0;                                     // stride hi | dim1_stride lo
    g1[7] = 0;
    i32x4 z4 = {0, 0, 0, 0};
#if __clang_major__ >= 23
    i32x8 z8 = {0, 0, 0, 0, 0, 0, 0, 0};
    __builtin_amdgcn_tensor_load_to_lds(g0, g1, z4, z4, z8, 0);
#else
    __builtin_amdgcn_tensor_load_to_lds(g0, g1, z4, z4, 0);
#endif
}

// ---------------------------------------------------------------------------
// Utility kernels
// ---------------------------------------------------------------------------
__global__ void zero_f32_kernel(float* __restrict__ p, long long n) {
    long long i = (long long)blockIdx.x * blockDim.x + threadIdx.x;
    long long stride = (long long)gridDim.x * blockDim.x;
    for (; i < n; i += stride) p[i] = 0.0f;
}

// mean-normalize rows: acc[r][c] /= (deg[r] + eps)
__global__ void div_deg_kernel(float* __restrict__ acc, const float* __restrict__ deg,
                               int R, int C) {
    long long i = (long long)blockIdx.x * blockDim.x + threadIdx.x;
    long long n = (long long)R * C;
    if (i >= n) return;
    acc[i] = acc[i] / (deg[(int)(i / C)] + EPSV);
}

// one-time weight prep: W[K x N] f32 -> Wt[N x K] bf16 (B-fragment friendly)
__global__ void wtrans_bf16_kernel(const float* __restrict__ W, ushort_t* __restrict__ Wt,
                                   int K, int N) {
    long long i = (long long)blockIdx.x * blockDim.x + threadIdx.x;
    if (i >= (long long)K * N) return;
    int k = (int)(i / N);
    int n = (int)(i - (long long)k * N);
    Wt[(size_t)n * K + k] = f32_to_bf16_rte(W[i]);
}

// ---------------------------------------------------------------------------
// Scatter-add kernels (segment_sum via fp32 atomics; targets are L2 resident)
// ---------------------------------------------------------------------------
// 768 floats per edge: one block per edge
__global__ __launch_bounds__(256)
void scatter_add_768_kernel(const float* __restrict__ X,
                            const int* __restrict__ srcIdx,
                            const int* __restrict__ dstIdx,
                            float* __restrict__ acc, float* __restrict__ deg, int E) {
    int e = blockIdx.x;
    if (e >= E) return;
    int s = srcIdx[e], d = dstIdx[e];
    const float* __restrict__ src = X + (size_t)s * 768;
    float* __restrict__ dst = acc + (size_t)d * 768;
    #pragma unroll
    for (int i = threadIdx.x; i < 768; i += 256)
        atomicAdd(&dst[i], src[i]);
    if (threadIdx.x == 0) atomicAdd(&deg[d], 1.0f);
}

// 128 floats per edge: one wave32 per edge (float4 per lane)
__global__ __launch_bounds__(256)
void scatter_add_128_kernel(const float* __restrict__ X,
                            const int* __restrict__ srcIdx,
                            const int* __restrict__ dstIdx,
                            float* __restrict__ acc, float* __restrict__ deg, int E) {
    int e = blockIdx.x * 8 + (threadIdx.x >> 5);
    if (e >= E) return;
    int lane = threadIdx.x & 31;
    int s = srcIdx[e], d = dstIdx[e];
    const float4 v = ((const float4*)(X + (size_t)s * 128))[lane];
    float* __restrict__ dst = acc + (size_t)d * 128 + lane * 4;
    atomicAdd(&dst[0], v.x);
    atomicAdd(&dst[1], v.y);
    atomicAdd(&dst[2], v.z);
    atomicAdd(&dst[3], v.w);
    if (lane == 0) atomicAdd(&deg[d], 1.0f);
}

// ---------------------------------------------------------------------------
// WMMA bf16 GEMM: Out = act(A @ W + bias)
//   A: [M x K] = concat(A0[M x KA0], A1[M x (K-KA0)]) f32, or bf16 (A0 only)
//   Wt: [N x K] bf16 (pre-transposed; L2 resident)
//   Block: 64 M-rows, FULL K staged once in LDS (CDNA5 320KB LDS/WGP).
//     - bf16 A: staged by the Tensor Data Mover (tensor_load_to_lds +
//       s_wait_tensorcnt), padding descriptor reproduces the K+8 LDS stride.
//     - f32 A: cooperative stage with on-the-fly bf16 conversion.
//   8 waves = 4 M-slices x 2 N-groups sweep all of N barrier-free; per n-chunk:
//   2 accumulators, K-loop of ds_b128 A-frags + global_b128 B-frags + 2 WMMA.
// ---------------------------------------------------------------------------
template<int K, int KA0, bool A_IS_BF16, bool TANH_BF16_OUT>
__global__ __launch_bounds__(256)
void gemm_wmma_kernel(const void* __restrict__ A0v, const void* __restrict__ A1v,
                      const ushort_t* __restrict__ Wt, const float* __restrict__ bias,
                      void* __restrict__ Outv, int M, int N) {
    constexpr int ROW = K + 8;             // padded LDS row (ushorts), 16B aligned
    __shared__ __align__(16) ushort_t As[64 * ROW];

    const int tid   = threadIdx.x;
    const int lane  = tid & 31;
    const int wave  = tid >> 5;
    const int mBase = blockIdx.x * 64;

    if (tid == 0) __builtin_prefetch(Wt, 0, 1);   // pull W stripe toward GL2

    if constexpr (A_IS_BF16) {
        // ---- TDM DMA: 64 x K bf16 tile, one descriptor per block ----
        constexpr int PADC = (K == 256) ? 6 : 7;  // 2^(c+1) DWORDs == one K row
        static_assert(K == 256 || K == 512, "TDM pad interval needs pow2 row");
        if (wave == 0) {
            unsigned lds_off = (unsigned)(unsigned long long)(const void*)&As[0];
            tdm_load_tile_bf16((const ushort_t*)A0v + (size_t)mBase * K,
                               lds_off, K, M - mBase, PADC);
            __builtin_amdgcn_s_wait_tensorcnt(0);
        }
        __syncthreads();
    } else {
        // ---- cooperative stage: 64 x K f32 -> bf16 LDS (once) ----
        constexpr int CPR = K / 8;                // 8-element chunks per row
        for (int c = tid; c < 64 * CPR; c += 256) {
            const int r  = c / CPR;
            const int kc = (c - r * CPR) * 8;
            const int m  = mBase + r;
            union { uint4 q; ushort_t u[8]; } pack;
            if (m < M) {
                // KA0 is a multiple of 8 -> each 8-chunk is source-uniform
                const float* src;
                if (kc < KA0) src = (const float*)A0v + (size_t)m * KA0 + kc;
                else          src = (const float*)A1v + (size_t)m * (K - KA0) + (kc - KA0);
                const float4 f0 = ((const float4*)src)[0];
                const float4 f1 = ((const float4*)src)[1];
                pack.u[0] = f32_to_bf16_rte(f0.x); pack.u[1] = f32_to_bf16_rte(f0.y);
                pack.u[2] = f32_to_bf16_rte(f0.z); pack.u[3] = f32_to_bf16_rte(f0.w);
                pack.u[4] = f32_to_bf16_rte(f1.x); pack.u[5] = f32_to_bf16_rte(f1.y);
                pack.u[6] = f32_to_bf16_rte(f1.z); pack.u[7] = f32_to_bf16_rte(f1.w);
            } else {
                pack.q.x = 0; pack.q.y = 0; pack.q.z = 0; pack.q.w = 0;
            }
            *(uint4*)&As[r * ROW + kc] = pack.q;
        }
        __syncthreads();
    }

    // ---- wave mapping: 4 M-slices x 2 N-groups ----
    const int wm = wave >> 1;                      // 0..3 -> 16-row slice
    const int wn = wave & 1;                       // 0..1 -> N interleave
    const int rowT = (wm << 4) + (lane & 15);
    const int kba  = (lane < 16) ? 0 : 8;          // A: lanes 16-31 hold K 8..15/24..31
    const ushort_t* __restrict__ aRow = &As[rowT * ROW];
    const int cT   = lane & 15;
    const int kbb  = (lane < 16) ? 0 : 8;          // B: lanes 16-31 hold K 8..15/24..31
    const int mrow = mBase + (wm << 4) + ((lane < 16) ? 0 : 8);

    union Frag { v16bf v; uint4 q[2]; };

    for (int n0 = wn * 32; n0 < N; n0 += 64) {
        v8f acc0 = {0.f,0.f,0.f,0.f,0.f,0.f,0.f,0.f};
        v8f acc1 = {0.f,0.f,0.f,0.f,0.f,0.f,0.f,0.f};
        const ushort_t* __restrict__ wt0 = Wt + (size_t)(n0 + cT)      * K + kbb;
        const ushort_t* __restrict__ wt1 = Wt + (size_t)(n0 + 16 + cT) * K + kbb;

        for (int k0 = 0; k0 < K; k0 += 32) {
            Frag fa, fb0, fb1;
            fa.q[0]  = *(const uint4*)(aRow + k0 + kba);
            fa.q[1]  = *(const uint4*)(aRow + k0 + kba + 16);
            fb0.q[0] = *(const uint4*)(wt0 + k0);
            fb0.q[1] = *(const uint4*)(wt0 + k0 + 16);
            fb1.q[0] = *(const uint4*)(wt1 + k0);
            fb1.q[1] = *(const uint4*)(wt1 + k0 + 16);
            acc0 = __builtin_amdgcn_wmma_f32_16x16x32_bf16(false, fa.v, false, fb0.v,
                                                           (short)0, acc0, false, false);
            acc1 = __builtin_amdgcn_wmma_f32_16x16x32_bf16(false, fa.v, false, fb1.v,
                                                           (short)0, acc1, false, false);
        }

        // epilogue: C layout lane L -> col L%16, VGPR v -> row v (+8 for L>=16)
        const float b0 = bias[n0 + cT];
        const float b1 = bias[n0 + 16 + cT];
        #pragma unroll
        for (int v = 0; v < 8; ++v) {
            const int m = mrow + v;
            if (m < M) {
                float x0 = acc0[v] + b0;
                float x1 = acc1[v] + b1;
                if constexpr (TANH_BF16_OUT) {
                    ushort_t* Out = (ushort_t*)Outv;
                    Out[(size_t)m * N + n0 + cT]      = f32_to_bf16_rte(tanhf(x0));
                    Out[(size_t)m * N + n0 + 16 + cT] = f32_to_bf16_rte(tanhf(x1));
                } else {
                    float* Out = (float*)Outv;
                    Out[(size_t)m * N + n0 + cT]      = x0;
                    Out[(size_t)m * N + n0 + 16 + cT] = x1;
                }
            }
        }
    }
}

// ---------------------------------------------------------------------------
// Host launcher
// ---------------------------------------------------------------------------
static inline size_t align256(size_t x) { return (x + 255) & ~(size_t)255; }

extern "C" void kernel_launch(void* const* d_in, const int* in_sizes, int n_in,
                              void* d_out, int out_size, void* d_ws, size_t ws_size,
                              hipStream_t stream) {
    const float* news_x  = (const float*)d_in[0];
    const float* users_x = (const float*)d_in[1];
    const float* w1a = (const float*)d_in[2];  const float* b1a = (const float*)d_in[3];
    const float* w1b = (const float*)d_in[4];  const float* b1b = (const float*)d_in[5];
    const float* wna = (const float*)d_in[6];  const float* bna = (const float*)d_in[7];
    const float* wnb = (const float*)d_in[8];  const float* bnb = (const float*)d_in[9];
    const float* wua = (const float*)d_in[10]; const float* bua = (const float*)d_in[11];
    const float* wub = (const float*)d_in[12]; const float* bub = (const float*)d_in[13];
    const int* atr_row = (const int*)d_in[14];
    const int* atr_col = (const int*)d_in[15];
    const int* usr_row = (const int*)d_in[16];
    const int* usr_col = (const int*)d_in[17];

    const int E_ATR = in_sizes[14];
    const int E_USR = in_sizes[16];
    const int NN = in_sizes[0] / 768;      // n_news  (50000)
    const int NU = in_sizes[1] / 128;      // n_users (100000)
    const int NA = 1000;                   // n_atr (reference constant)

    // ---- workspace layout ----
    // contiguous f32 accumulator region (zeroed each call):
    float* base_f = (float*)d_ws;
    size_t off = 0;
    float* deg_atr  = base_f + off; off += NA;
    float* agr_atr  = base_f + off; off += (size_t)NA * 768;
    float* deg_news = base_f + off; off += NN;
    float* agr_news = base_f + off; off += (size_t)NN * 128;
    float* deg_usr  = base_f + off; off += NU;
    float* agr_usr  = base_f + off; off += (size_t)NU * 128;
    const long long acc_floats = (long long)off;

    // bf16 activation / f32 feature region
    size_t boff = align256(off * sizeof(float));
    ushort_t* h1 = (ushort_t*)((char*)d_ws + boff);            // [NA,256] bf16
    boff = align256(boff + (size_t)NA * 256 * sizeof(ushort_t));
    float* atr_feats = (float*)((char*)d_ws + boff);           // [NA,128] f32
    boff = align256(boff + (size_t)NA * 128 * sizeof(float));
    ushort_t* hn = (ushort_t*)((char*)d_ws + boff);            // [NN,512] bf16
    boff = align256(boff + (size_t)NN * 512 * sizeof(ushort_t));
    ushort_t* hu = (ushort_t*)((char*)d_ws + boff);            // [NU,256] bf16
    boff = align256(boff + (size_t)NU * 256 * sizeof(ushort_t));
    // pre-transposed bf16 weights Wt[N][K]
    ushort_t* w1aT = (ushort_t*)((char*)d_ws + boff); boff = align256(boff + (size_t)768*256*2);
    ushort_t* w1bT = (ushort_t*)((char*)d_ws + boff); boff = align256(boff + (size_t)256*128*2);
    ushort_t* wnaT = (ushort_t*)((char*)d_ws + boff); boff = align256(boff + (size_t)896*512*2);
    ushort_t* wnbT = (ushort_t*)((char*)d_ws + boff); boff = align256(boff + (size_t)512*128*2);
    ushort_t* wuaT = (ushort_t*)((char*)d_ws + boff); boff = align256(boff + (size_t)256*256*2);
    ushort_t* wubT = (ushort_t*)((char*)d_ws + boff); boff = align256(boff + (size_t)256*128*2);

    float* news_out  = (float*)d_out;                          // [NN,128]
    float* users_out = (float*)d_out + (size_t)NN * 128;       // [NU,128]

    // 0) zero scatter accumulators + prep bf16 transposed weights
    zero_f32_kernel<<<4096, 256, 0, stream>>>(base_f, acc_floats);
    wtrans_bf16_kernel<<<(768*256 + 255)/256, 256, 0, stream>>>(w1a, w1aT, 768, 256);
    wtrans_bf16_kernel<<<(256*128 + 255)/256, 256, 0, stream>>>(w1b, w1bT, 256, 128);
    wtrans_bf16_kernel<<<(896*512 + 255)/256, 256, 0, stream>>>(wna, wnaT, 896, 512);
    wtrans_bf16_kernel<<<(512*128 + 255)/256, 256, 0, stream>>>(wnb, wnbT, 512, 128);
    wtrans_bf16_kernel<<<(256*256 + 255)/256, 256, 0, stream>>>(wua, wuaT, 256, 256);
    wtrans_bf16_kernel<<<(256*128 + 255)/256, 256, 0, stream>>>(wub, wubT, 256, 128);

    // 1) FirstConv scatter: news -> attribute nodes (mean)
    scatter_add_768_kernel<<<E_ATR, 256, 0, stream>>>(news_x, atr_row, atr_col,
                                                      agr_atr, deg_atr, E_ATR);
    div_deg_kernel<<<((long long)NA * 768 + 255) / 256, 256, 0, stream>>>(
        agr_atr, deg_atr, NA, 768);

    // 2) attribute MLP
    gemm_wmma_kernel<768, 768, false, true><<<(NA + 63) / 64, 256, 0, stream>>>(
        agr_atr, nullptr, w1aT, b1a, h1, NA, 256);
    gemm_wmma_kernel<256, 256, true, false><<<(NA + 63) / 64, 256, 0, stream>>>(
        h1, nullptr, w1bT, b1b, atr_feats, NA, 128);

    // 3) NewsConv scatter: attribute -> news (mean)
    scatter_add_128_kernel<<<(E_ATR + 7) / 8, 256, 0, stream>>>(
        atr_feats, atr_col, atr_row, agr_news, deg_news, E_ATR);
    div_deg_kernel<<<((long long)NN * 128 + 255) / 256, 256, 0, stream>>>(
        agr_news, deg_news, NN, 128);

    // 4) news MLP on concat(news_x[768], agr_news[128])
    gemm_wmma_kernel<896, 768, false, true><<<(NN + 63) / 64, 256, 0, stream>>>(
        news_x, agr_news, wnaT, bna, hn, NN, 512);
    gemm_wmma_kernel<512, 512, true, false><<<(NN + 63) / 64, 256, 0, stream>>>(
        hn, nullptr, wnbT, bnb, news_out, NN, 128);

    // 5) UserConv scatter: news_out -> users (mean); gathers straight from d_out
    scatter_add_128_kernel<<<(E_USR + 7) / 8, 256, 0, stream>>>(
        news_out, usr_row, usr_col, agr_usr, deg_usr, E_USR);
    div_deg_kernel<<<((long long)NU * 128 + 255) / 256, 256, 0, stream>>>(
        agr_usr, deg_usr, NU, 128);

    // 6) user MLP on concat(users_x[128], agr_usr[128])
    gemm_wmma_kernel<256, 128, false, true><<<(NU + 63) / 64, 256, 0, stream>>>(
        users_x, agr_usr, wuaT, bua, hu, NU, 256);
    gemm_wmma_kernel<256, 256, true, false><<<(NU + 63) / 64, 256, 0, stream>>>(
        hu, nullptr, wubT, bub, users_out, NU, 128);
}
